// Attention_7035156431415
// MI455X (gfx1250) — matrix operand order repaired and checked by reading, not compiled, not run on hardware
//
#include <hip/hip_runtime.h>
#include <hip/hip_bf16.h>

// ---------------------------------------------------------------------------
// Problem constants (match reference)
// ---------------------------------------------------------------------------
#define B_    2
#define N_    2048
#define DIM_  1024
#define H_    16
#define DH_   64
#define SCALE_ 8.0f
#define ROWS_ (B_ * N_)          // 4096
#define QKVN_ (3 * DIM_)         // 3072

#define TILE_M 64
#define TILE_N 256
#define TILE_K 64                // K-slice staged per barrier pair

typedef __bf16 bf16_t;
typedef __attribute__((ext_vector_type(16))) __bf16 v16bf;
typedef __attribute__((ext_vector_type(8)))  __bf16 v8bf;
typedef __attribute__((ext_vector_type(4)))  __bf16 v4bf;
typedef __attribute__((ext_vector_type(8)))  float  v8f;
typedef __attribute__((ext_vector_type(4)))  unsigned int v4u;
typedef __attribute__((ext_vector_type(4)))  int  v4i;
typedef __attribute__((ext_vector_type(8)))  int  v8i;

#if __has_builtin(__builtin_amdgcn_tensor_load_to_lds) && \
    __has_builtin(__builtin_amdgcn_s_wait_tensorcnt)
#define USE_TDM 1
#else
#define USE_TDM 0
#endif

static __device__ __forceinline__ v16bf cat8(v8bf lo, v8bf hi) {
  return __builtin_shufflevector(lo, hi, 0,1,2,3,4,5,6,7,8,9,10,11,12,13,14,15);
}

// A-fragment 16x32 bf16 (per ISA table): lane -> row m = lane&15,
// halves 0..7 = K (lane>>4)*8 .. +7, halves 8..15 = K 16+(lane>>4)*8 .. +7.
static __device__ __forceinline__ v16bf load_a_frag(const bf16_t* __restrict__ row_base,
                                                    int ldk, int kk, int lane) {
  const int m = lane & 15, koff = (lane >> 4) * 8;
  const bf16_t* p = row_base + (size_t)m * ldk + kk + koff;
  v8bf lo = *(const v8bf*)(p);
  v8bf hi = *(const v8bf*)(p + 16);
  return cat8(lo, hi);
}

// B-fragment 32x16 bf16: lane -> col n = lane&15,
// halves 0..7 = K (lane>>4)*8 .. +7, halves 8..15 = K 16+(lane>>4)*8 .. +7.
static __device__ __forceinline__ v16bf load_b_frag(const bf16_t* __restrict__ col_base,
                                                    int ldk, int kk, int lane) {
  const int n = lane & 15, koff = (lane >> 4) * 8;
  const bf16_t* p = col_base + (size_t)n * ldk + kk + koff;
  v8bf lo = *(const v8bf*)(p);
  v8bf hi = *(const v8bf*)(p + 16);
  return cat8(lo, hi);
}

static __device__ __forceinline__ v8f wmma_bf16(v16bf a, v16bf b, v8f c) {
  return __builtin_amdgcn_wmma_f32_16x16x32_bf16(false, a, false, b, (short)0, c, false, false);
}

static __device__ __forceinline__ float wave_sum32(float v) {
  #pragma unroll
  for (int off = 16; off > 0; off >>= 1) v += __shfl_xor(v, off);
  return v;
}

#if USE_TDM
// ---------------------------------------------------------------------------
// Tensor Data Mover: DMA a 2D bf16 tile [tile_rows x tile_cols] (row stride
// row_stride elems, columns contiguous) from global into LDS (rows packed).
// D# per CDNA5 ISA 8.3/8.4.  Wave-level op (EXEC ignored); TENSORcnt-tracked.
// ---------------------------------------------------------------------------
static __device__ __forceinline__ void tdm_load_tile(bf16_t* lds_dst, const bf16_t* gsrc,
                                                     int tile_cols, int tile_rows,
                                                     int row_stride,
                                                     int tensor_rows, int tensor_cols) {
  const unsigned lds_addr = (unsigned)(unsigned long long)lds_dst;  // flat LDS addr[31:0]
  const unsigned long long ga = (unsigned long long)gsrc;
  v4u g0;
  g0[0] = 1u;                                            // count=1 (valid), user mode
  g0[1] = lds_addr;                                      // lds_addr
  g0[2] = (unsigned)ga;                                  // global_addr lo
  g0[3] = ((unsigned)(ga >> 32) & 0x01FFFFFFu) | (2u << 30);  // addr[56:32] | type=2
  v8i g1;
  g1[0] = (int)(1u << 16);                               // workgroup_mask=0, data_size=1 (2B)
  g1[1] = (int)(((unsigned)tensor_cols & 0xFFFFu) << 16);        // tensor_dim0[15:0]
  g1[2] = (int)((((unsigned)tensor_cols >> 16) & 0xFFFFu) |
                (((unsigned)tensor_rows & 0xFFFFu) << 16));      // dim0 hi | dim1 lo
  g1[3] = (int)((((unsigned)tensor_rows >> 16) & 0xFFFFu) |
                (((unsigned)tile_cols & 0xFFFFu) << 16));        // dim1 hi | tile_dim0
  g1[4] = (int)((unsigned)tile_rows & 0xFFFFu);                  // tile_dim1 (tile_dim2=0)
  g1[5] = (int)(unsigned)row_stride;                             // tensor_dim0_stride lo
  g1[6] = 0;
  g1[7] = 0;
  v4i zz = (v4i){0, 0, 0, 0};
#if __clang_major__ >= 23
  v8i z8 = (v8i){0, 0, 0, 0, 0, 0, 0, 0};
  __builtin_amdgcn_tensor_load_to_lds(g0, g1, zz, zz, z8, 0);
#else
  __builtin_amdgcn_tensor_load_to_lds(g0, g1, zz, zz, 0);
#endif
}
#endif  // USE_TDM

// ---------------------------------------------------------------------------
// K0a: fp32 [R][C] -> bf16 [C][R] (K-major weights)
// ---------------------------------------------------------------------------
__global__ __launch_bounds__(256)
void transpose_to_bf16_kernel(const float* __restrict__ in, bf16_t* __restrict__ out,
                              int R, int C) {
  size_t t = (size_t)blockIdx.x * blockDim.x + threadIdx.x;
  if (t >= (size_t)R * C) return;
  int r = (int)(t / C), c = (int)(t % C);
  out[(size_t)c * R + r] = (bf16_t)in[t];
}

// K0b: flat fp32 -> bf16
__global__ __launch_bounds__(256)
void convert_bf16_kernel(const float* __restrict__ in, bf16_t* __restrict__ out, size_t n) {
  size_t t = (size_t)blockIdx.x * blockDim.x + threadIdx.x;
  if (t < n) out[t] = (bf16_t)in[t];
}

// ---------------------------------------------------------------------------
// K1: LayerNorm, one wave per 1024-wide row, fp32 stats, bf16 output
// ---------------------------------------------------------------------------
__global__ __launch_bounds__(256)
void layernorm_kernel(const float* __restrict__ x, const float* __restrict__ g,
                      const float* __restrict__ bb, bf16_t* __restrict__ y, int rows) {
  const int lane = threadIdx.x & 31, wid = threadIdx.x >> 5;
  const int row = blockIdx.x * 8 + wid;
  if (row >= rows) return;
  const float4* xr = (const float4*)(x + (size_t)row * DIM_);
  float4 v[8];
  float s = 0.f, sq = 0.f;
  #pragma unroll
  for (int e = 0; e < 8; e++) {
    v[e] = xr[e * 32 + lane];
    s  += v[e].x + v[e].y + v[e].z + v[e].w;
    sq += v[e].x * v[e].x + v[e].y * v[e].y + v[e].z * v[e].z + v[e].w * v[e].w;
  }
  s  = wave_sum32(s);
  sq = wave_sum32(sq);
  const float mu   = s * (1.f / DIM_);
  const float var  = sq * (1.f / DIM_) - mu * mu;
  const float rstd = rsqrtf(var + 1e-5f);
  #pragma unroll
  for (int e = 0; e < 8; e++) {
    const int d = (e * 32 + lane) * 4;
    v4bf o;
    o[0] = (bf16_t)((v[e].x - mu) * rstd * g[d + 0] + bb[d + 0]);
    o[1] = (bf16_t)((v[e].y - mu) * rstd * g[d + 1] + bb[d + 1]);
    o[2] = (bf16_t)((v[e].z - mu) * rstd * g[d + 2] + bb[d + 2]);
    o[3] = (bf16_t)((v[e].w - mu) * rstd * g[d + 3] + bb[d + 3]);
    *(v4bf*)(y + (size_t)row * DIM_ + d) = o;
  }
}

// ---------------------------------------------------------------------------
// WMMA GEMM  C[M][N] = A[M][K](bf16,row-major) x BT[N][K](bf16,K-major)
// (+bias[N]).  Block = 8 waves = 64x256 tile; wave = 32x64 = 2x4 fragments.
// 64-wide K slices staged in LDS by the TDM (double-buffered, wave 0 issues,
// TENSORcnt + split barriers); 16 WMMAs + 24 ds_load_b128 per barrier pair.
// ---------------------------------------------------------------------------
__global__ __launch_bounds__(256)
void gemm_bf16_kernel(const bf16_t* __restrict__ A, const bf16_t* __restrict__ BT,
                      const float* __restrict__ bias,
                      float* __restrict__ Cf, bf16_t* __restrict__ Cb,
                      int M, int N, int K) {
  __shared__ __align__(16) bf16_t lA[2][TILE_M * TILE_K];   // 16 KB
  __shared__ __align__(16) bf16_t lB[2][TILE_N * TILE_K];   // 64 KB

  const int lane = threadIdx.x & 31;
  const int wid  = threadIdx.x >> 5;
  const int wm = wid & 1, wn = wid >> 1;
  const int m0 = blockIdx.x * TILE_M;
  const int n0 = blockIdx.y * TILE_N;
  const bf16_t* Abase = A + (size_t)m0 * K;
  const bf16_t* Bbase = BT + (size_t)n0 * K;

  v8f acc[2][4];
  #pragma unroll
  for (int i = 0; i < 2; i++)
    #pragma unroll
    for (int j = 0; j < 4; j++)
      acc[i][j] = (v8f){0.f,0.f,0.f,0.f,0.f,0.f,0.f,0.f};

#if USE_TDM
  if (wid == 0) {
    tdm_load_tile(&lA[0][0], Abase, TILE_K, TILE_M, K, M, K);
    tdm_load_tile(&lB[0][0], Bbase, TILE_K, TILE_N, K, N, K);
  }
#endif

  for (int kk = 0; kk < K; kk += TILE_K) {
#if USE_TDM
    const int cur = (kk / TILE_K) & 1;
    const bool more = (kk + TILE_K) < K;
    if (wid == 0) {
      if (more) {   // prior tail barrier makes buf[cur^1] safe to overwrite
        tdm_load_tile(&lA[cur ^ 1][0], Abase + kk + TILE_K, TILE_K, TILE_M, K, M, K);
        tdm_load_tile(&lB[cur ^ 1][0], Bbase + kk + TILE_K, TILE_K, TILE_N, K, N, K);
        __builtin_amdgcn_s_wait_tensorcnt(2);   // first 2 (current buffer) done
      } else {
        __builtin_amdgcn_s_wait_tensorcnt(0);
      }
    }
    __syncthreads();                            // tile visible to all waves
#else
    const int cur = 0;
    __syncthreads();
    {  // cooperative synchronous staging fallback (16B chunks)
      const int tid = threadIdx.x;
      #pragma unroll
      for (int e = 0; e < (TILE_M * TILE_K) / (256 * 8); e++) {
        int idx = (tid + e * 256) * 8;
        int r = idx / TILE_K, c = idx % TILE_K;
        *(v8bf*)(&lA[0][idx]) = *(const v8bf*)(Abase + (size_t)r * K + kk + c);
      }
      #pragma unroll
      for (int e = 0; e < (TILE_N * TILE_K) / (256 * 8); e++) {
        int idx = (tid + e * 256) * 8;
        int r = idx / TILE_K, c = idx % TILE_K;
        *(v8bf*)(&lB[0][idx]) = *(const v8bf*)(Bbase + (size_t)r * K + kk + c);
      }
    }
    __syncthreads();
#endif

    #pragma unroll
    for (int ks = 0; ks < TILE_K; ks += 32) {
      v16bf a0 = load_a_frag(&lA[cur][(wm * 32) * TILE_K],      TILE_K, ks, lane);
      v16bf a1 = load_a_frag(&lA[cur][(wm * 32 + 16) * TILE_K], TILE_K, ks, lane);
      #pragma unroll
      for (int j = 0; j < 4; j++) {
        v16bf bj = load_b_frag(&lB[cur][(wn * 64 + 16 * j) * TILE_K], TILE_K, ks, lane);
        acc[0][j] = wmma_bf16(a0, bj, acc[0][j]);
        acc[1][j] = wmma_bf16(a1, bj, acc[1][j]);
      }
    }
#if USE_TDM
    __syncthreads();                            // all reads done before next overwrite
#endif
  }

  const int half = lane >> 4, ln = lane & 15;
  #pragma unroll
  for (int i = 0; i < 2; i++) {
    #pragma unroll
    for (int j = 0; j < 4; j++) {
      const int col = n0 + wn * 64 + 16 * j + ln;
      const float bv = bias ? bias[col] : 0.f;
      #pragma unroll
      for (int r = 0; r < 8; r++) {
        const int row = m0 + wm * 32 + 16 * i + r + 8 * half;
        const float v = acc[i][j][r] + bv;
        if (Cf) Cf[(size_t)row * N + col] = v;
        if (Cb) Cb[(size_t)row * N + col] = (bf16_t)v;
      }
    }
  }
}

// ---------------------------------------------------------------------------
// K3: split heads, L2-normalize q,k -> bf16 [b,h,n,dh]; v -> bf16 transposed
//     [b,h,dh,n].  One wave per (b,h,n) row.
// ---------------------------------------------------------------------------
__global__ __launch_bounds__(256)
void qkv_head_kernel(const float* __restrict__ qkv, bf16_t* __restrict__ qn,
                     bf16_t* __restrict__ kn, bf16_t* __restrict__ vt) {
  const int lane = threadIdx.x & 31, wid = threadIdx.x >> 5;
  const int rid = blockIdx.x * 8 + wid;        // (b*H + h)*N + n
  const int n  = rid % N_;
  const int bh = rid / N_;
  const int h  = bh % H_;
  const int b  = bh / H_;
  const float* base = qkv + (size_t)(b * N_ + n) * QKVN_ + h * DH_;
  const float2 q2 = *(const float2*)(base + 2 * lane);
  const float2 k2 = *(const float2*)(base + DIM_ + 2 * lane);
  const float2 v2 = *(const float2*)(base + 2 * DIM_ + 2 * lane);
  const float qs = wave_sum32(q2.x * q2.x + q2.y * q2.y);
  const float ks = wave_sum32(k2.x * k2.x + k2.y * k2.y);
  const float qi = 1.f / fmaxf(sqrtf(qs), 1e-12f);
  const float ki = 1.f / fmaxf(sqrtf(ks), 1e-12f);
  const size_t ro = (size_t)rid * DH_ + 2 * lane;
  qn[ro]     = (bf16_t)(q2.x * qi);
  qn[ro + 1] = (bf16_t)(q2.y * qi);
  kn[ro]     = (bf16_t)(k2.x * ki);
  kn[ro + 1] = (bf16_t)(k2.y * ki);
  const size_t vo = ((size_t)bh * DH_ + 2 * lane) * N_ + n;
  vt[vo]      = (bf16_t)v2.x;
  vt[vo + N_] = (bf16_t)v2.y;
}

// ---------------------------------------------------------------------------
// K4: causal flash attention.  Block = 4 waves sharing one (b,h) and a
// 64-query super-tile; wave w owns queries [i0_blk+16w, +16).  K/V tiles for
// 32 keys are TDM-staged into double-buffered LDS and shared by all 4 waves.
// Fully-masked steps for low waves are numerically benign (p -> 0, alpha=1).
// ---------------------------------------------------------------------------
__global__ __launch_bounds__(128)
void flash_attn_kernel(const bf16_t* __restrict__ qn, const bf16_t* __restrict__ kn,
                       const bf16_t* __restrict__ vt, bf16_t* __restrict__ o) {
  __shared__ __align__(16) bf16_t lK[2][32 * DH_];   // 8 KB  [key][dh]
  __shared__ __align__(16) bf16_t lV[2][DH_ * 32];   // 8 KB  [dh][key]
  __shared__ __align__(16) bf16_t plds[4][16 * 32];  // 4 KB  per-wave P tiles

  const int lane = threadIdx.x & 31, wid = threadIdx.x >> 5;
  const int qb = blockIdx.x & (N_ / 64 - 1);   // 64-query block index
  const int bh = blockIdx.x / (N_ / 64);
  const int h = bh % H_, b = bh / H_;
  const int i0_blk = qb * 64;
  const int i0 = i0_blk + wid * 16;
  const int half = lane >> 4, ln = lane & 15;

  const bf16_t* qbase = qn + ((size_t)bh * N_ + i0) * DH_;
  const bf16_t* kbase = kn + (size_t)bh * N_ * DH_;
  const bf16_t* vbase = vt + (size_t)bh * DH_ * N_;

  v16bf qa0 = load_a_frag(qbase, DH_, 0, lane);
  v16bf qa1 = load_a_frag(qbase, DH_, 32, lane);

  v8f o_acc[4];
  #pragma unroll
  for (int dt = 0; dt < 4; dt++) o_acc[dt] = (v8f){0.f,0.f,0.f,0.f,0.f,0.f,0.f,0.f};
  float mrow[8], lrow[8];
  #pragma unroll
  for (int r = 0; r < 8; r++) { mrow[r] = -1e30f; lrow[r] = 0.f; }

  const int jsteps = (i0_blk + 64) / 32;       // keys 0 .. i0_blk+63

#if USE_TDM
  if (wid == 0) {
    tdm_load_tile(&lK[0][0], kbase, DH_, 32, DH_, N_, DH_);          // 32 keys x 64 dh
    tdm_load_tile(&lV[0][0], vbase, 32, DH_, N_, DH_, N_);           // 64 dh x 32 keys
  }
#endif

  for (int st = 0; st < jsteps; ++st) {
    const int j0 = st * 32;
#if USE_TDM
    const int cur = st & 1;
    const bool more = (st + 1) < jsteps;
    if (wid == 0) {
      if (more) {
        tdm_load_tile(&lK[cur ^ 1][0], kbase + (size_t)(j0 + 32) * DH_, DH_, 32, DH_, N_, DH_);
        tdm_load_tile(&lV[cur ^ 1][0], vbase + j0 + 32, 32, DH_, N_, DH_, N_);
        __builtin_amdgcn_s_wait_tensorcnt(2);
      } else {
        __builtin_amdgcn_s_wait_tensorcnt(0);
      }
    }
    __syncthreads();
#else
    const int cur = 0;
    __syncthreads();
    {  // synchronous staging fallback
      const int tid = threadIdx.x;
      #pragma unroll
      for (int e = 0; e < (32 * DH_) / (128 * 8); e++) {
        int idx = (tid + e * 128) * 8;
        *(v8bf*)(&lK[0][idx]) = *(const v8bf*)(kbase + (size_t)j0 * DH_ + idx);
      }
      #pragma unroll
      for (int e = 0; e < (DH_ * 32) / (128 * 8); e++) {
        int idx = (tid + e * 128) * 8;
        int r = idx / 32, c = idx % 32;
        *(v8bf*)(&lV[0][idx]) = *(const v8bf*)(vbase + (size_t)r * N_ + j0 + c);
      }
    }
    __syncthreads();
#endif

    v8f s[2];
    #pragma unroll
    for (int t = 0; t < 2; t++) {
      s[t] = (v8f){0.f,0.f,0.f,0.f,0.f,0.f,0.f,0.f};
      v16bf b0 = load_b_frag(&lK[cur][(16 * t) * DH_], DH_, 0, lane);
      v16bf b1 = load_b_frag(&lK[cur][(16 * t) * DH_], DH_, 32, lane);
      s[t] = wmma_bf16(qa0, b0, s[t]);
      s[t] = wmma_bf16(qa1, b1, s[t]);
    }
    #pragma unroll
    for (int r = 0; r < 8; r++) {
      const int q = i0 + r + 8 * half;
      float t0 = s[0][r] * SCALE_;
      float t1 = s[1][r] * SCALE_;
      if (j0 + ln > q)      t0 = -1e30f;
      if (j0 + 16 + ln > q) t1 = -1e30f;
      float mx = fmaxf(t0, t1);
      #pragma unroll
      for (int off = 1; off < 16; off <<= 1) mx = fmaxf(mx, __shfl_xor(mx, off));
      const float mn = fmaxf(mrow[r], mx);
      const float p0 = __expf(t0 - mn);
      const float p1 = __expf(t1 - mn);
      const float al = __expf(mrow[r] - mn);
      float rs = p0 + p1;
      #pragma unroll
      for (int off = 1; off < 16; off <<= 1) rs += __shfl_xor(rs, off);
      lrow[r] = lrow[r] * al + rs;
      mrow[r] = mn;
      plds[wid][(r + 8 * half) * 32 + ln]      = (bf16_t)p0;
      plds[wid][(r + 8 * half) * 32 + 16 + ln] = (bf16_t)p1;
      #pragma unroll
      for (int dt = 0; dt < 4; dt++) o_acc[dt][r] *= al;
    }
    v16bf pf = load_a_frag(&plds[wid][0], 32, 0, lane);   // C-layout -> A-fragment
    #pragma unroll
    for (int dt = 0; dt < 4; dt++) {
      v16bf vb = load_b_frag(&lV[cur][(dt * 16) * 32], 32, 0, lane);
      o_acc[dt] = wmma_bf16(pf, vb, o_acc[dt]);
    }
    __syncthreads();                            // readers done before next overwrite
  }

  #pragma unroll
  for (int r = 0; r < 8; r++) {
    const float inv = 1.f / lrow[r];
    const int q = i0 + r + 8 * half;
    #pragma unroll
    for (int dt = 0; dt < 4; dt++)
      o[((size_t)b * N_ + q) * DIM_ + h * DH_ + dt * 16 + ln] = (bf16_t)(o_acc[dt][r] * inv);
  }
}

// ---------------------------------------------------------------------------
// Host launcher
// ---------------------------------------------------------------------------
extern "C" void kernel_launch(void* const* d_in, const int* in_sizes, int n_in,
                              void* d_out, int out_size, void* d_ws, size_t ws_size,
                              hipStream_t stream) {
  const float* x        = (const float*)d_in[0];
  const float* g        = (const float*)d_in[1];
  const float* bparam   = (const float*)d_in[2];
  const float* c_attn_w = (const float*)d_in[3];
  const float* c_attn_b = (const float*)d_in[4];
  const float* c_proj_w = (const float*)d_in[5];
  const float* c_proj_b = (const float*)d_in[6];
  const float* to_out_w = (const float*)d_in[7];

  char* ws = (char*)d_ws;
  size_t off = 0;
  auto alloc = [&](size_t bytes) -> void* {
    void* p = ws + off;
    off += (bytes + 255) & ~(size_t)255;
    return p;
  };
  bf16_t* y_b    = (bf16_t*)alloc((size_t)ROWS_ * DIM_ * 2);       // reused for o_b
  bf16_t* wqkvT  = (bf16_t*)alloc((size_t)QKVN_ * DIM_ * 2);
  bf16_t* wprojT = (bf16_t*)alloc((size_t)DIM_ * DIM_ * 2);
  bf16_t* wout   = (bf16_t*)alloc((size_t)DIM_ * DIM_ * 2);
  float*  qkv_f  = (float*)alloc((size_t)ROWS_ * QKVN_ * 4);
  bf16_t* qn     = (bf16_t*)alloc((size_t)B_ * H_ * N_ * DH_ * 2); // reused for proj_b
  bf16_t* kn     = (bf16_t*)alloc((size_t)B_ * H_ * N_ * DH_ * 2);
  bf16_t* vt     = (bf16_t*)alloc((size_t)B_ * H_ * N_ * DH_ * 2);
  bf16_t* o_b    = y_b;    // y dead after GEMM1
  bf16_t* proj_b = qn;     // qn dead after flash attention

  {
    size_t n1 = (size_t)DIM_ * QKVN_;
    transpose_to_bf16_kernel<<<dim3((unsigned)((n1 + 255) / 256)), dim3(256), 0, stream>>>(
        c_attn_w, wqkvT, DIM_, QKVN_);
    size_t n2 = (size_t)DIM_ * DIM_;
    transpose_to_bf16_kernel<<<dim3((unsigned)((n2 + 255) / 256)), dim3(256), 0, stream>>>(
        c_proj_w, wprojT, DIM_, DIM_);
    convert_bf16_kernel<<<dim3((unsigned)((n2 + 255) / 256)), dim3(256), 0, stream>>>(
        to_out_w, wout, n2);
  }

  layernorm_kernel<<<dim3(ROWS_ / 8), dim3(256), 0, stream>>>(x, g, bparam, y_b, ROWS_);

  gemm_bf16_kernel<<<dim3(ROWS_ / TILE_M, QKVN_ / TILE_N), dim3(256), 0, stream>>>(
      y_b, wqkvT, c_attn_b, qkv_f, (bf16_t*)nullptr, ROWS_, QKVN_, DIM_);

  qkv_head_kernel<<<dim3(B_ * H_ * N_ / 8), dim3(256), 0, stream>>>(qkv_f, qn, kn, vt);

  flash_attn_kernel<<<dim3(B_ * H_ * (N_ / 64)), dim3(128), 0, stream>>>(qn, kn, vt, o_b);

  gemm_bf16_kernel<<<dim3(ROWS_ / TILE_M, DIM_ / TILE_N), dim3(256), 0, stream>>>(
      o_b, wprojT, c_proj_b, (float*)nullptr, proj_b, ROWS_, DIM_, DIM_);

  gemm_bf16_kernel<<<dim3(ROWS_ / TILE_M, DIM_ / TILE_N), dim3(256), 0, stream>>>(
      proj_b, wout, (const float*)nullptr, (float*)d_out, (bf16_t*)nullptr, ROWS_, DIM_, DIM_);
}
